// GatedGraphConv_91302414778944
// MI455X (gfx1250) — matrix operand order repaired and checked by reading, not compiled, not run on hardware
//
#include <hip/hip_runtime.h>
#include <hip/hip_bf16.h>

typedef __attribute__((ext_vector_type(16))) __bf16 v16bf;
typedef __attribute__((ext_vector_type(8)))  __bf16 bf16x8;
typedef __attribute__((ext_vector_type(4)))  __bf16 bf16x4;
typedef __attribute__((ext_vector_type(8)))  float  v8f;

#define H 128        // hidden size
#define G3 384       // 3*H
#define MT 32        // node rows per block tile (2 WMMA M-tiles)

// ---------------- small utility kernels ----------------

__global__ void cvt_bf16_kernel(const float* __restrict__ s, __bf16* __restrict__ d, int n) {
    int i = blockIdx.x * blockDim.x + threadIdx.x;
    if (i < n) d[i] = (__bf16)s[i];
}

__global__ void zero_f32_kernel(float* __restrict__ p, int n) {
    int i = blockIdx.x * blockDim.x + threadIdx.x;
    if (i < n) p[i] = 0.0f;
}

// h = pad(features, 64 -> 128); write f32 + bf16 copies
__global__ void init_h_kernel(const float* __restrict__ feat,
                              float* __restrict__ hf, __bf16* __restrict__ hb,
                              int n_nodes) {
    int i = blockIdx.x * blockDim.x + threadIdx.x;
    if (i >= n_nodes * H) return;
    int node = i >> 7, c = i & (H - 1);
    float v = (c < 64) ? feat[node * 64 + c] : 0.0f;
    hf[i] = v;
    hb[i] = (__bf16)v;
}

__global__ void deg_kernel(const int* __restrict__ dst, float* __restrict__ deg, int n_edges) {
    int i = blockIdx.x * blockDim.x + threadIdx.x;
    if (i < n_edges) unsafeAtomicAdd(&deg[dst[i]], 1.0f);
}

// one wave (32 lanes) per edge: agg[dst] += (float)h_bf16[src]
__global__ __launch_bounds__(256) void scatter_kernel(
    const __bf16* __restrict__ hb, const int* __restrict__ src,
    const int* __restrict__ dst, float* __restrict__ agg, int n_edges) {
    int gid  = blockIdx.x * blockDim.x + threadIdx.x;
    int edge = gid >> 5;
    int lane = gid & 31;
    if (edge >= n_edges) return;
    int s = src[edge], d = dst[edge];
    const __bf16* hp = hb  + (size_t)s * H + lane * 4;
    float*        ap = agg + (size_t)d * H + lane * 4;
    bf16x4 v = *(const bf16x4*)hp;
    #pragma unroll
    for (int j = 0; j < 4; ++j) unsafeAtomicAdd(ap + j, (float)v[j]);
}

// ---------------- WMMA helpers (16x16x32 bf16, f32 accum) ----------------

// A fragment (16x32, MxK) from row-major LDS tile [16][128] at k-base kb.
// Lane L: row = L%16; elems 0..15 = K kb+(L/16)*16 .. +15
__device__ inline v16bf load_a_frag(const __bf16* base, int kb, int lane) {
    int row  = lane & 15;
    int half = lane >> 4;
    const __bf16* r = base + row * H + kb + half * 16;
    return *(const v16bf*)(r);
}

// B fragment (32x16, KxN) for X@W^T with row-major W[outdim][128] bf16.
// Lane L holds column n = n0 + L%16, K = kb + (L/16)*16 .. +15 -> one 32B row chunk.
__device__ inline v16bf load_b_frag(const __bf16* __restrict__ Wb, int n0, int kb, int lane) {
    int n    = n0 + (lane & 15);
    int koff = kb + ((lane >> 4) << 4);
    return *(const v16bf*)(Wb + (size_t)n * H + koff);
}

__device__ inline v8f wmma_bf16(v16bf a, v16bf b, v8f c) {
    return __builtin_amdgcn_wmma_f32_16x16x32_bf16(false, a, false, b, (short)0, c, false, false);
}

__device__ inline float fast_sigmoid(float x) { return 1.0f / (1.0f + __expf(-x)); }

// ---------------- fused per-step kernel ----------------
// Per 32-node tile (256 threads = 8 waves):
//   phase A: a = agg_bf16 @ W^T + deg*b     (wave w -> n-tile w; 2 M-tiles share B frag)
//   phase B: wave w -> column patch w*16 of the GRU; computes all 3 gates of
//            gi = a@W_ih^T and gh = h@W_hh^T for both M-tiles (12 accumulators),
//            then does the GRU elementwise directly in WMMA C-layout registers.
__global__ __launch_bounds__(256) void gru_step_kernel(
    const float* __restrict__ agg, __bf16* __restrict__ hb,
    float* __restrict__ hf,
    const __bf16* __restrict__ Wb,   const float* __restrict__ bvec,
    const float* __restrict__ deg,
    const __bf16* __restrict__ Wihb, const __bf16* __restrict__ Whhb,
    const float* __restrict__ bih,   const float* __restrict__ bhh,
    int n_nodes) {
    __shared__ __bf16 sAgg[MT * H];   // 8KB
    __shared__ __bf16 sA  [MT * H];   // 8KB
    __shared__ __bf16 sH  [MT * H];   // 8KB
    __shared__ float  sDeg[MT];

    int tid  = threadIdx.x;
    int lane = tid & 31;
    int wave = tid >> 5;                 // 0..7
    int row0 = blockIdx.x * MT;
    if (row0 >= n_nodes) return;
    int last = n_nodes - 1;

    // ---- load tile: agg (f32->bf16) and h (bf16) into LDS ----
    {
        int r = tid >> 3, c = (tid & 7) * 16;   // 256 threads: 32 rows x 8 chunks of 16
        int node = row0 + r; if (node > last) node = last;
        size_t gbase = (size_t)node * H + c;
        const float* ap = agg + gbase;
        #pragma unroll
        for (int j = 0; j < 16; ++j) sAgg[r * H + c + j] = (__bf16)ap[j];
        const __bf16* hp = hb + gbase;
        *(bf16x8*)&sH[r * H + c]     = *(const bf16x8*)(hp);
        *(bf16x8*)&sH[r * H + c + 8] = *(const bf16x8*)(hp + 8);
        if (tid < MT) {
            int dnode = row0 + tid; if (dnode > last) dnode = last;
            sDeg[tid] = deg[dnode];
        }
    }
    __syncthreads();

    // ---- phase A: a = agg @ W^T + deg*b -> sA (bf16); wave w owns n-tile w ----
    {
        int t = wave;                        // 8 n-tiles of width 16
        v8f acc0 = {}, acc1 = {};
        #pragma unroll
        for (int kt = 0; kt < 4; ++kt) {
            v16bf bf = load_b_frag(Wb, t * 16, kt * 32, lane);
            v16bf a0 = load_a_frag(sAgg,            kt * 32, lane);
            v16bf a1 = load_a_frag(sAgg + 16 * H,   kt * 32, lane);
            acc0 = wmma_bf16(a0, bf, acc0);
            acc1 = wmma_bf16(a1, bf, acc1);
        }
        int n = t * 16 + (lane & 15);
        float bn = bvec[n];
        int mbase = (lane >> 4) * 8;         // C layout: row = mbase+v, col = n
        #pragma unroll
        for (int v = 0; v < 8; ++v) {
            int r0 = mbase + v, r1 = 16 + mbase + v;
            sA[r0 * H + n] = (__bf16)(acc0[v] + sDeg[r0] * bn);
            sA[r1 * H + n] = (__bf16)(acc1[v] + sDeg[r1] * bn);
        }
    }
    __syncthreads();

    // ---- phase B: wave w owns output columns [w*16, w*16+16) ----
    {
        int ct = wave;                        // column patch
        v8f ai[2][3], ah[2][3];
        #pragma unroll
        for (int m = 0; m < 2; ++m)
            #pragma unroll
            for (int g = 0; g < 3; ++g) { ai[m][g] = (v8f){}; ah[m][g] = (v8f){}; }

        #pragma unroll
        for (int kt = 0; kt < 4; ++kt) {
            v16bf aA0 = load_a_frag(sA,          kt * 32, lane);
            v16bf aA1 = load_a_frag(sA + 16 * H, kt * 32, lane);
            v16bf aH0 = load_a_frag(sH,          kt * 32, lane);
            v16bf aH1 = load_a_frag(sH + 16 * H, kt * 32, lane);
            #pragma unroll
            for (int g = 0; g < 3; ++g) {
                v16bf bI = load_b_frag(Wihb, g * H + ct * 16, kt * 32, lane);
                ai[0][g] = wmma_bf16(aA0, bI, ai[0][g]);
                ai[1][g] = wmma_bf16(aA1, bI, ai[1][g]);
                v16bf bH = load_b_frag(Whhb, g * H + ct * 16, kt * 32, lane);
                ah[0][g] = wmma_bf16(aH0, bH, ah[0][g]);
                ah[1][g] = wmma_bf16(aH1, bH, ah[1][g]);
            }
        }

        // GRU elementwise directly in C-layout registers
        int n  = ct * 16 + (lane & 15);       // output column 0..127
        float bir = bih[n],           bhr = bhh[n];
        float biz = bih[H + n],       bhz = bhh[H + n];
        float bin = bih[2 * H + n],   bhn = bhh[2 * H + n];
        int mbase = (lane >> 4) * 8;
        #pragma unroll
        for (int m = 0; m < 2; ++m) {
            #pragma unroll
            for (int v = 0; v < 8; ++v) {
                int row  = m * 16 + mbase + v;
                int node = row0 + row;
                if (node >= n_nodes) continue;
                size_t gi = (size_t)node * H + n;
                float rg = fast_sigmoid((ai[m][0][v] + bir) + (ah[m][0][v] + bhr));
                float zg = fast_sigmoid((ai[m][1][v] + biz) + (ah[m][1][v] + bhz));
                float ng = tanhf((ai[m][2][v] + bin) + rg * (ah[m][2][v] + bhn));
                float ho = hf[gi];
                float hn = (1.0f - zg) * ng + zg * ho;
                hf[gi] = hn;
                hb[gi] = (__bf16)hn;
            }
        }
    }
}

// ---------------- host launcher ----------------

extern "C" void kernel_launch(void* const* d_in, const int* in_sizes, int n_in,
                              void* d_out, int out_size, void* d_ws, size_t ws_size,
                              hipStream_t stream) {
    const float* feat = (const float*)d_in[0];   // [N,64]
    const float* W    = (const float*)d_in[1];   // [128,128]
    const float* b    = (const float*)d_in[2];   // [128]
    const float* Wih  = (const float*)d_in[3];   // [384,128]
    const float* Whh  = (const float*)d_in[4];   // [384,128]
    const float* bih  = (const float*)d_in[5];   // [384]
    const float* bhh  = (const float*)d_in[6];   // [384]
    const int*   src  = (const int*)d_in[7];     // [E]
    const int*   dst  = (const int*)d_in[8];     // [E]
    int n_nodes = in_sizes[0] / 64;
    int n_edges = in_sizes[7];
    float* hf = (float*)d_out;                   // h lives in d_out (f32)

    // workspace carve-out (256B aligned)
    char* ws = (char*)d_ws;
    size_t off = 0;
    auto carve = [&](size_t bytes) -> char* {
        char* p = ws + off;
        off = (off + bytes + 255) & ~(size_t)255;
        return p;
    };
    __bf16* hb   = (__bf16*)carve((size_t)n_nodes * H * 2);
    float*  agg  = (float*) carve((size_t)n_nodes * H * 4);
    float*  deg  = (float*) carve((size_t)n_nodes * 4);
    __bf16* Wb   = (__bf16*)carve((size_t)H  * H * 2);
    __bf16* Wihb = (__bf16*)carve((size_t)G3 * H * 2);
    __bf16* Whhb = (__bf16*)carve((size_t)G3 * H * 2);

    const int BT = 256;
    cvt_bf16_kernel<<<(H * H  + BT - 1) / BT, BT, 0, stream>>>(W,   Wb,   H * H);
    cvt_bf16_kernel<<<(G3 * H + BT - 1) / BT, BT, 0, stream>>>(Wih, Wihb, G3 * H);
    cvt_bf16_kernel<<<(G3 * H + BT - 1) / BT, BT, 0, stream>>>(Whh, Whhb, G3 * H);

    int nh = n_nodes * H;
    init_h_kernel<<<(nh + BT - 1) / BT, BT, 0, stream>>>(feat, hf, hb, n_nodes);

    zero_f32_kernel<<<(n_nodes + BT - 1) / BT, BT, 0, stream>>>(deg, n_nodes);
    deg_kernel<<<(n_edges + BT - 1) / BT, BT, 0, stream>>>(dst, deg, n_edges);

    for (int step = 0; step < 3; ++step) {
        zero_f32_kernel<<<(nh + BT - 1) / BT, BT, 0, stream>>>(agg, nh);
        scatter_kernel<<<((n_edges * 32) + BT - 1) / BT, BT, 0, stream>>>(hb, src, dst, agg, n_edges);
        gru_step_kernel<<<(n_nodes + MT - 1) / MT, 256, 0, stream>>>(
            agg, hb, hf, Wb, b, deg, Wihb, Whhb, bih, bhh, n_nodes);
    }
}
